// TransformerTranslator_64604898066462
// MI455X (gfx1250) — matrix operand order repaired and checked by reading, not compiled, not run on hardware
//
#include <hip/hip_runtime.h>
#include <hip/hip_bf16.h>

typedef __bf16 bf16;
typedef __attribute__((ext_vector_type(16))) __bf16 v16bf;
typedef __attribute__((ext_vector_type(8)))  float  v8f;

#define NB    256
#define TT    43
#define HH    128
#define NA    2
#define DKV   96
#define DFF_  2048
#define VOUT_ 10000
#define NTOK  (NB*TT)          // 11008 = 86*128
#define NQKV  (3*NA*DKV)       // 576
#define VPAD  10048            // 10000 padded to multiple of 64

// ---------------- embed: x = emb[tok] + pos[t] ----------------
__global__ void k_embed(const int* __restrict__ tok, const float* __restrict__ emb,
                        const float* __restrict__ pos, float* __restrict__ xf,
                        bf16* __restrict__ xb) {
    int row = blockIdx.x;            // 0..NTOK-1
    int h   = threadIdx.x;           // 0..127
    int t   = row % TT;
    int id  = tok[row];
    float v = emb[(size_t)id * HH + h] + pos[t * HH + h];
    xf[(size_t)row * HH + h] = v;
    xb[(size_t)row * HH + h] = (bf16)v;
}

// ------- pack row-major fp32 W[K,N] -> bf16 B-fragment layout (pad N) -------
// packed index = (((ntile*KB + kb)*32) + lane)*16 + e ; n = ntile*16 + lane%16,
// k = kb*32 + (lane/16)*16 + e   (contiguous k order, same as the A loads in k_gemm)
__global__ void k_pack_b(const float* __restrict__ W, bf16* __restrict__ out,
                         int K, int Nsrc, int Npad) {
    int tid = blockIdx.x * 256 + threadIdx.x;
    int KB = K >> 5;
    if (tid >= Npad * K) return;
    int e    = tid & 15;
    int lane = (tid >> 4) & 31;
    int rest = tid >> 9;
    int kb   = rest % KB;
    int nt   = rest / KB;
    int n    = nt * 16 + (lane & 15);
    int k    = kb * 32 + (lane >> 4) * 16 + e;
    float v  = (n < Nsrc) ? W[(size_t)k * Nsrc + n] : 0.f;
    out[tid] = (bf16)v;
}

// ------- pack Wq/Wk/Wv (A,H,DK) into one fused [128 x 576] B + fused bias -------
__global__ void k_pack_qkv(const float* __restrict__ Wq, const float* __restrict__ Wk,
                           const float* __restrict__ Wv, const float* __restrict__ bq,
                           const float* __restrict__ bk, const float* __restrict__ bv,
                           bf16* __restrict__ out, float* __restrict__ bias) {
    int tid = blockIdx.x * 256 + threadIdx.x;
    const int K = HH, KB = HH >> 5;
    if (tid >= NQKV * K) return;
    int e    = tid & 15;
    int lane = (tid >> 4) & 31;
    int rest = tid >> 9;
    int kb   = rest % KB;
    int nt   = rest / KB;
    int n    = nt * 16 + (lane & 15);
    int k    = kb * 32 + (lane >> 4) * 16 + e;
    int sec  = n / (NA * DKV);            // 0=Q 1=K 2=V
    int w    = n % (NA * DKV);
    int a    = w / DKV, d = w % DKV;
    const float* W = (sec == 0) ? Wq : (sec == 1) ? Wk : Wv;
    out[tid] = (bf16)W[((size_t)a * HH + k) * DKV + d];
    if (tid < NQKV) {
        int sec2 = tid / (NA * DKV), w2 = tid % (NA * DKV);
        const float* B = (sec2 == 0) ? bq : (sec2 == 1) ? bk : bv;
        bias[tid] = B[w2];
    }
}

// ---------------- bf16 WMMA GEMM: C[M,Nreal] = A[M,K] @ Bpacked + bias ----------------
// 8 waves/block; each wave owns a 32Mx32N tile (2x2 WMMA tiles, fragment reuse).
// Block tile: 128M x 64N.
union FragB16 { uint4 u[2]; v16bf v; };

template<int BIAS, int RELU, int OUTBF16>
__global__ __launch_bounds__(256) void k_gemm(
        const bf16* __restrict__ A, const bf16* __restrict__ Bp,
        const float* __restrict__ bias, void* __restrict__ C,
        int M, int K, int Nreal) {
    int lane = threadIdx.x & 31;
    int wave = threadIdx.x >> 5;
    int wm = wave >> 1, wn = wave & 1;
    int tileM  = blockIdx.x * 128 + wm * 32;     // wave rows [tileM, tileM+32)
    int ntile0 = blockIdx.y * 4 + wn * 2;        // wave owns 16-wide n-tiles ntile0, ntile0+1
    int KB     = K >> 5;
    int half   = lane >> 4;

    const bf16* Arow0 = A + (size_t)(tileM + (lane & 15)) * K + half * 16;
    const bf16* Arow1 = Arow0 + (size_t)16 * K;
    const bf16* Bl0   = Bp + ((size_t)ntile0 * KB * 32 + lane) * 16;
    const bf16* Bl1   = Bl0 + (size_t)KB * 32 * 16;

    v8f acc[2][2] = {};
    for (int kb = 0; kb < KB; ++kb) {
        FragB16 fa0, fa1, fb0, fb1;
        fa0.u[0] = *(const uint4*)(Arow0);  fa0.u[1] = *(const uint4*)(Arow0 + 8);
        fa1.u[0] = *(const uint4*)(Arow1);  fa1.u[1] = *(const uint4*)(Arow1 + 8);
        fb0.u[0] = *(const uint4*)(Bl0);    fb0.u[1] = *(const uint4*)(Bl0 + 8);
        fb1.u[0] = *(const uint4*)(Bl1);    fb1.u[1] = *(const uint4*)(Bl1 + 8);
        Arow0 += 32; Arow1 += 32; Bl0 += 512; Bl1 += 512;
        __builtin_prefetch(Arow0, 0, 1);     // global_prefetch_b8 next fragments
        __builtin_prefetch(Arow1, 0, 1);
        __builtin_prefetch(Bl0, 0, 1);
        __builtin_prefetch(Bl1, 0, 1);
        acc[0][0] = __builtin_amdgcn_wmma_f32_16x16x32_bf16(
                        false, fa0.v, false, fb0.v, (short)0, acc[0][0], false, false);
        acc[0][1] = __builtin_amdgcn_wmma_f32_16x16x32_bf16(
                        false, fa0.v, false, fb1.v, (short)0, acc[0][1], false, false);
        acc[1][0] = __builtin_amdgcn_wmma_f32_16x16x32_bf16(
                        false, fa1.v, false, fb0.v, (short)0, acc[1][0], false, false);
        acc[1][1] = __builtin_amdgcn_wmma_f32_16x16x32_bf16(
                        false, fa1.v, false, fb1.v, (short)0, acc[1][1], false, false);
    }

    int nl = lane & 15;
#pragma unroll
    for (int tn = 0; tn < 2; ++tn) {
        int n = (ntile0 + tn) * 16 + nl;
        if (n >= Nreal) continue;
        float bv_ = BIAS ? bias[n] : 0.f;
#pragma unroll
        for (int tm = 0; tm < 2; ++tm) {
            int mbase = tileM + tm * 16 + 8 * half;
#pragma unroll
            for (int r = 0; r < 8; ++r) {
                float cv = acc[tm][tn][r] + bv_;
                if (RELU) cv = fmaxf(cv, 0.f);
                size_t idx = (size_t)(mbase + r) * Nreal + n;
                if (OUTBF16) ((bf16*)C)[idx] = (bf16)cv;
                else         ((float*)C)[idx] = cv;
            }
        }
    }
}

// ---------------- attention: one wave per (n, head, t) ----------------
__global__ void k_attn(const float* __restrict__ qkv, bf16* __restrict__ multi) {
    int b = blockIdx.x;                    // 0 .. NB*NA*TT-1
    int t = b % TT;
    int tmp = b / TT;
    int a = tmp % NA;
    int n = tmp / NA;
    int lane = threadIdx.x;                // 0..31 (wave32)

    const float* q = qkv + (size_t)(n * TT + t) * NQKV + a * DKV;
    float s0 = 0.f, s1 = 0.f;
    int sB = lane + 32;
    {
        const float* kp = qkv + (size_t)(n * TT + lane) * NQKV + NA * DKV + a * DKV;
        for (int d = 0; d < DKV; ++d) s0 += q[d] * kp[d];
    }
    if (sB < TT) {
        const float* kp = qkv + (size_t)(n * TT + sB) * NQKV + NA * DKV + a * DKV;
        for (int d = 0; d < DKV; ++d) s1 += q[d] * kp[d];
    }
    const float scale = 0.1020620726f;     // 1/sqrt(96)
    s0 *= scale; s1 *= scale;
    float mx = (sB < TT) ? fmaxf(s0, s1) : s0;
    for (int off = 16; off; off >>= 1) mx = fmaxf(mx, __shfl_xor(mx, off, 32));
    float p0 = __expf(s0 - mx);
    float p1 = (sB < TT) ? __expf(s1 - mx) : 0.f;
    float sum = p0 + p1;
    for (int off = 16; off; off >>= 1) sum += __shfl_xor(sum, off, 32);
    float inv = 1.f / sum;

    __shared__ float pr[64];
    pr[lane]      = p0 * inv;
    pr[lane + 32] = p1 * inv;              // 0 when invalid
    __syncthreads();

    for (int dv = lane; dv < DKV; dv += 32) {
        float accv = 0.f;
        for (int s = 0; s < TT; ++s)
            accv += pr[s] * qkv[(size_t)(n * TT + s) * NQKV + 2 * NA * DKV + a * DKV + dv];
        multi[(size_t)(n * TT + t) * (NA * DKV) + a * DKV + dv] = (bf16)accv;
    }
}

// ---------------- residual add + LayerNorm (one row / 128-thread block) ----------------
__global__ void k_add_ln(const float* __restrict__ xa, const float* __restrict__ xr,
                         const float* __restrict__ g, const float* __restrict__ be,
                         float* __restrict__ of, bf16* __restrict__ ob) {
    int row = blockIdx.x;
    int h   = threadIdx.x;                 // 0..127
    __shared__ float red[128];
    float v = xa[(size_t)row * HH + h] + xr[(size_t)row * HH + h];
    red[h] = v; __syncthreads();
    for (int off = 64; off; off >>= 1) { if (h < off) red[h] += red[h + off]; __syncthreads(); }
    float mean = red[0] * (1.f / HH);
    __syncthreads();
    float d = v - mean;
    red[h] = d * d; __syncthreads();
    for (int off = 64; off; off >>= 1) { if (h < off) red[h] += red[h + off]; __syncthreads(); }
    float var = red[0] * (1.f / HH);
    float y = d * __frsqrt_rn(var + 1e-5f) * g[h] + be[h];
    of[(size_t)row * HH + h] = y;
    ob[(size_t)row * HH + h] = (bf16)y;
}

// ---------------- host side ----------------
extern "C" void kernel_launch(void* const* d_in, const int* in_sizes, int n_in,
                              void* d_out, int out_size, void* d_ws, size_t ws_size,
                              hipStream_t stream) {
    const int*   tok  = (const int*)  d_in[0];
    const float* emb  = (const float*)d_in[1];
    const float* pos  = (const float*)d_in[2];
    const float* Wq   = (const float*)d_in[3];
    const float* bq   = (const float*)d_in[4];
    const float* Wk   = (const float*)d_in[5];
    const float* bk   = (const float*)d_in[6];
    const float* Wv   = (const float*)d_in[7];
    const float* bv   = (const float*)d_in[8];
    const float* Wo   = (const float*)d_in[9];
    const float* bo   = (const float*)d_in[10];
    const float* ln1g = (const float*)d_in[11];
    const float* ln1b = (const float*)d_in[12];
    const float* W1   = (const float*)d_in[13];
    const float* b1   = (const float*)d_in[14];
    const float* W2   = (const float*)d_in[15];
    const float* b2   = (const float*)d_in[16];
    const float* ln2g = (const float*)d_in[17];
    const float* ln2b = (const float*)d_in[18];
    const float* Wout = (const float*)d_in[19];
    const float* bout = (const float*)d_in[20];
    float* out = (float*)d_out;

    char* p = (char*)d_ws;
    auto alloc = [&](size_t bytes) -> char* {
        char* r = p; p += (bytes + 255) & ~(size_t)255; return r;
    };
    float* xf     = (float*)alloc((size_t)NTOK * HH   * 4);
    bf16*  xb     = (bf16*) alloc((size_t)NTOK * HH   * 2);
    float* qkvf   = (float*)alloc((size_t)NTOK * NQKV * 4);
    bf16*  multib = (bf16*) alloc((size_t)NTOK * NA * DKV * 2);
    float* attnf  = (float*)alloc((size_t)NTOK * HH   * 4);
    float* x1f    = (float*)alloc((size_t)NTOK * HH   * 4);
    bf16*  x1b    = (bf16*) alloc((size_t)NTOK * HH   * 2);
    bf16*  hb     = (bf16*) alloc((size_t)NTOK * DFF_ * 2);
    float* fff    = (float*)alloc((size_t)NTOK * HH   * 4);
    float* x2f    = (float*)alloc((size_t)NTOK * HH   * 4);
    bf16*  x2b    = (bf16*) alloc((size_t)NTOK * HH   * 2);
    bf16*  Bqkv   = (bf16*) alloc((size_t)HH * NQKV * 2);
    float* qkvbias= (float*)alloc(NQKV * 4);
    bf16*  Bo     = (bf16*) alloc((size_t)(NA*DKV) * HH * 2);
    bf16*  B1     = (bf16*) alloc((size_t)HH * DFF_ * 2);
    bf16*  B2     = (bf16*) alloc((size_t)DFF_ * HH * 2);
    bf16*  Bvout  = (bf16*) alloc((size_t)HH * VPAD * 2);

    // weight packing
    k_pack_qkv<<<(NQKV * HH + 255) / 256, 256, 0, stream>>>(Wq, Wk, Wv, bq, bk, bv, Bqkv, qkvbias);
    k_pack_b<<<((NA*DKV) * HH + 255) / 256, 256, 0, stream>>>(Wo,   Bo,    NA*DKV, HH,    HH);
    k_pack_b<<<(HH * DFF_     + 255) / 256, 256, 0, stream>>>(W1,   B1,    HH,     DFF_,  DFF_);
    k_pack_b<<<(DFF_ * HH     + 255) / 256, 256, 0, stream>>>(W2,   B2,    DFF_,   HH,    HH);
    k_pack_b<<<(HH * VPAD     + 255) / 256, 256, 0, stream>>>(Wout, Bvout, HH,     VOUT_, VPAD);

    // embed
    k_embed<<<NTOK, HH, 0, stream>>>(tok, emb, pos, xf, xb);

    const int MB = NTOK / 128;  // 86
    // fused QKV projection
    k_gemm<1,0,0><<<dim3(MB, NQKV/64), 256, 0, stream>>>(xb, Bqkv, qkvbias, qkvf, NTOK, HH, NQKV);
    // attention
    k_attn<<<NB * NA * TT, 32, 0, stream>>>(qkvf, multib);
    // output projection
    k_gemm<1,0,0><<<dim3(MB, HH/64), 256, 0, stream>>>(multib, Bo, bo, attnf, NTOK, NA*DKV, HH);
    // residual + LN1
    k_add_ln<<<NTOK, HH, 0, stream>>>(xf, attnf, ln1g, ln1b, x1f, x1b);
    // FFN up (bias + relu, bf16 out)
    k_gemm<1,1,1><<<dim3(MB, DFF_/64), 256, 0, stream>>>(x1b, B1, b1, hb, NTOK, HH, DFF_);
    // FFN down
    k_gemm<1,0,0><<<dim3(MB, HH/64), 256, 0, stream>>>(hb, B2, b2, fff, NTOK, DFF_, HH);
    // residual + LN2
    k_add_ln<<<NTOK, HH, 0, stream>>>(x1f, fff, ln2g, ln2b, x2f, x2b);
    // vocab projection straight into d_out
    k_gemm<1,0,0><<<dim3(MB, VPAD/64), 256, 0, stream>>>(x2b, Bvout, bout, out, NTOK, HH, VOUT_);
}
